// GNNModule_52080773431961
// MI455X (gfx1250) — compile-verified, matrix operand order checked by reading, not run on hardware
//
#include <hip/hip_runtime.h>
#include <hip/hip_bf16.h>
#include <stdint.h>

// Problem constants (from reference)
#define NNODES 100000
#define NEDGES 3200000
#define FIN    256
#define KTOT   1024      // concat K: x | hop0 | hop1 | hop2
#define COUT   512       // concat out: alpha(256) | beta(256)
#define BM     128
#define BN     128
#define BK     32
#define LDSPAD 8         // pad K-stride to 40 bf16 (80B) to stagger banks

// TDM availability (device pass only; host pass parses the fallback path)
#if defined(__gfx1250__) && defined(__has_builtin)
#  if __has_builtin(__builtin_amdgcn_tensor_load_to_lds)
#    define HAVE_TDM 1
#  endif
#endif
#ifndef HAVE_TDM
#  define HAVE_TDM 0
#endif
// amdgpu-toolchain (therock headers) ships the TDM header and uses the 6-arg builtin;
// ROCm 7.2 has the 5-arg form.
#if __has_include(<hip/amd_detail/amd_gfx1250_TDM.h>)
#  define TDM_6ARG 1
#else
#  define TDM_6ARG 0
#endif

typedef __attribute__((ext_vector_type(16))) __bf16        v16bf;
typedef __attribute__((ext_vector_type(8)))  float         v8f;
typedef __attribute__((ext_vector_type(4)))  unsigned int  uintx4;
typedef __attribute__((ext_vector_type(8)))  int           intx8;
typedef __attribute__((ext_vector_type(4)))  int           intx4;

union FragU { uint4 q[2]; v16bf v; };

// Load one 16x32 bf16 fragment per the CDNA5 16-bit A/B layout:
// lane L in [0,16): row = base+L, VGPR0..3 = K 0..7, VGPR4..7 = K 16..23
// lane L in [16,32): row = base+L-16, VGPR0..3 = K 8..15, VGPR4..7 = K 24..31
__device__ __forceinline__ v16bf load_frag(const __bf16* rowPtr, int kh) {
    FragU f;
    f.q[0] = *(const uint4*)(rowPtr + kh);        // 8 bf16 = 16B
    f.q[1] = *(const uint4*)(rowPtr + kh + 16);   // 8 bf16 = 16B
    return f.v;
}

__device__ __forceinline__ unsigned int pack_bf2(float a, float b) {
    union { __bf16 h[2]; unsigned int u; } x;
    x.h[0] = (__bf16)a;
    x.h[1] = (__bf16)b;
    return x.u;
}

#if HAVE_TDM
// Issue one TDM 2D tile load: 128 rows (N, stride 1024 elems) x 32 elems (K,
// contiguous) of bf16 from gsrc (tile start) into LDS at lds_off.
// LDS padding: 16B pad every 64B stored -> 80B row stride == BK+LDSPAD bf16.
__device__ __forceinline__ void tdm_load_b_tile(const __bf16* gsrc, unsigned int lds_off) {
    const unsigned long long ga = (unsigned long long)(uintptr_t)gsrc;
    uintx4 g0;
    g0.x = 1u;                                              // count=1, user mode
    g0.y = lds_off;                                         // lds_addr [63:32]
    g0.z = (unsigned int)ga;                                // global_addr[31:0]
    g0.w = (unsigned int)((ga >> 32) & 0x01FFFFFFu)         // global_addr[56:32]
         | 0x80000000u;                                     // type=2 (bits 127:126)
    intx8 g1;
    g1[0] = (int)((1u << 16)          // data_size = 2 bytes
                | (1u << 20)          // pad_enable
                | (3u << 22)          // pad_interval: 16 DWORDs = 64B
                | (3u << 25));        // pad_amount: 4 DWORDs = 16B
    g1[1] = (int)(1024u << 16);       // tensor_dim0 = 1024 (K), bits 79:48
    g1[2] = (int)(512u << 16);        // tensor_dim1 = 512 (N), bits 111:80
    g1[3] = (int)(32u << 16);         // tile_dim0 = 32 (K), bits 127:112
    g1[4] = 128;                      // tile_dim1 = 128 (N); tile_dim2 = 0
    g1[5] = 1024;                     // tensor_dim0_stride = 1024 elems (low 32)
    g1[6] = 0;                        // stride hi; tensor_dim1_stride lo
    g1[7] = 0;
    const intx4 z4 = {0, 0, 0, 0};    // 2D tensor: groups 2/3 unused
#if TDM_6ARG
    const intx8 z8 = {0, 0, 0, 0, 0, 0, 0, 0};
    __builtin_amdgcn_tensor_load_to_lds(g0, g1, z4, z4, z8, 0);
#else
    __builtin_amdgcn_tensor_load_to_lds(g0, g1, z4, z4, 0);
#endif
}
#endif

// ---------------------------------------------------------------------------
// Weight pack + BN constant fold.
// Wt[n][k] (bf16, transposed): n in [0,512), k in [0,1024).
// ---------------------------------------------------------------------------
__global__ void prep_kernel(const float* __restrict__ wa_x, const float* __restrict__ wa_adj,
                            const float* __restrict__ wb_x, const float* __restrict__ wb_adj,
                            const float* __restrict__ bnag, const float* __restrict__ bnab,
                            const float* __restrict__ bnam, const float* __restrict__ bnav,
                            const float* __restrict__ bnbg, const float* __restrict__ bnbb,
                            const float* __restrict__ bnbm, const float* __restrict__ bnbv,
                            __bf16* __restrict__ Wt,
                            float* __restrict__ bnscale, float* __restrict__ bnshift) {
    const int n = blockIdx.x;                          // 0..511 output column
    const int k = blockIdx.y * 256 + threadIdx.x;      // 0..1023 input feature
    const int c = n & 255;
    const bool isA = (n < 256);
    const float* wx   = isA ? wa_x   : wb_x;
    const float* wadj = isA ? wa_adj : wb_adj;
    const int g = k >> 8;
    const int r = k & 255;
    float w;
    if (g == 0)
        w = wx[r * 256 + c] + wx[65536 + r * 256 + c] + wx[131072 + r * 256 + c];
    else
        w = wadj[(g - 1) * 65536 + r * 256 + c];
    Wt[(size_t)n * KTOT + k] = (__bf16)w;

    if (k == 0) {
        const float gmm = isA ? bnag[c] : bnbg[c];
        const float bet = isA ? bnab[c] : bnbb[c];
        const float mu  = isA ? bnam[c] : bnbm[c];
        const float var = isA ? bnav[c] : bnbv[c];
        const float s = gmm * rsqrtf(var + 1e-3f);
        bnscale[n] = s;
        bnshift[n] = bet - mu * s;
    }
}

// ---------------------------------------------------------------------------
// SpMM: y[row] += val * h[col], one wave per edge, 8 features per lane.
// Gathers mostly hit the 192MB L2 (hop matrix = 102MB); scatter via relaxed
// agent-scope f32 atomics (non-returning global_atomic_add_f32).
// ---------------------------------------------------------------------------
__global__ void spmm_kernel(const float* __restrict__ hin, float* __restrict__ hout,
                            const int* __restrict__ erow, const int* __restrict__ ecol,
                            const float* __restrict__ eval) {
    const int e = blockIdx.x * 8 + (threadIdx.x >> 5);
    if (e >= NEDGES) return;
    const int lane = threadIdx.x & 31;
    const int r = erow[e];
    const int c = ecol[e];
    const float v = eval[e];
    const float4* pin = (const float4*)(hin + (size_t)c * FIN + lane * 8);
    const float4 a = pin[0];
    const float4 b = pin[1];
    float* po = hout + (size_t)r * FIN + lane * 8;
    __hip_atomic_fetch_add(po + 0, v * a.x, __ATOMIC_RELAXED, __HIP_MEMORY_SCOPE_AGENT);
    __hip_atomic_fetch_add(po + 1, v * a.y, __ATOMIC_RELAXED, __HIP_MEMORY_SCOPE_AGENT);
    __hip_atomic_fetch_add(po + 2, v * a.z, __ATOMIC_RELAXED, __HIP_MEMORY_SCOPE_AGENT);
    __hip_atomic_fetch_add(po + 3, v * a.w, __ATOMIC_RELAXED, __HIP_MEMORY_SCOPE_AGENT);
    __hip_atomic_fetch_add(po + 4, v * b.x, __ATOMIC_RELAXED, __HIP_MEMORY_SCOPE_AGENT);
    __hip_atomic_fetch_add(po + 5, v * b.y, __ATOMIC_RELAXED, __HIP_MEMORY_SCOPE_AGENT);
    __hip_atomic_fetch_add(po + 6, v * b.z, __ATOMIC_RELAXED, __HIP_MEMORY_SCOPE_AGENT);
    __hip_atomic_fetch_add(po + 7, v * b.w, __ATOMIC_RELAXED, __HIP_MEMORY_SCOPE_AGENT);
}

// ---------------------------------------------------------------------------
// Fused GEMM + BN epilogue:
//   out[N,512] = BN( [x|hop0|hop1|hop2] (N,1024) @ Wt^T (1024,512) )
// 128x128 block tile, 8 waves, wave = 64x32 subtile = 8 v_wmma accumulators.
// A staged fp32->bf16 into LDS by VALU; B tile DMA'd by the Tensor Data Mover.
// ---------------------------------------------------------------------------
__global__ void __launch_bounds__(256)
gemm_kernel(const float* __restrict__ x,  const float* __restrict__ h0,
            const float* __restrict__ h1, const float* __restrict__ h2,
            const __bf16* __restrict__ Wt,
            const float* __restrict__ bnscale, const float* __restrict__ bnshift,
            float* __restrict__ out) {
    __shared__ __align__(16) __bf16 As[BM][BK + LDSPAD];
    __shared__ __align__(16) __bf16 Bs[BN][BK + LDSPAD];

    const int t    = threadIdx.x;
    const int lane = t & 31;
    const int wave = t >> 5;
    const int rowBlock = blockIdx.x * BM;
    const int colBlock = blockIdx.y * BN;
    const int mBase = (wave >> 2) * 64;     // 2 row-groups of 64
    const int nBase = (wave & 3) * 32;      // 4 col-groups of 32
    const int lr = lane & 15;
    const int kh = (lane >> 4) * 8;

    v8f acc[4][2] = {};

    const int sRow = t >> 1;                // 0..127: LDS row staged by this thread
    const int sK   = (t & 1) * 16;          // K half within the 32-wide tile
    const int gRow = rowBlock + sRow;
    const bool rowOk = (gRow < NNODES);

#if HAVE_TDM
    const unsigned int bs_lds = (unsigned int)(uintptr_t)(void*)&Bs[0][0];
#endif
    const __bf16* WtCol = Wt + (size_t)colBlock * KTOT;

    const float* const srcs[4] = {x, h0, h1, h2};
#pragma unroll
    for (int g = 0; g < 4; ++g) {
        const float* const src = srcs[g];          // constant per unrolled copy
        const float* const srcRow = src + (size_t)gRow * FIN + sK;
        for (int kk = 0; kk < 256; kk += BK) {
            const int k0 = g * 256 + kk;

            // ---- stage A tile: 128 rows x 32 k, fp32 -> bf16 ----
            {
                unsigned int* d = (unsigned int*)&As[sRow][sK];
                if (rowOk) {
                    const float4* p = (const float4*)(srcRow + kk);
                    const float4 v0 = p[0], v1 = p[1], v2 = p[2], v3 = p[3];
                    d[0] = pack_bf2(v0.x, v0.y);
                    d[1] = pack_bf2(v0.z, v0.w);
                    d[2] = pack_bf2(v1.x, v1.y);
                    d[3] = pack_bf2(v1.z, v1.w);
                    d[4] = pack_bf2(v2.x, v2.y);
                    d[5] = pack_bf2(v2.z, v2.w);
                    d[6] = pack_bf2(v3.x, v3.y);
                    d[7] = pack_bf2(v3.z, v3.w);
                } else {
#pragma unroll
                    for (int i = 0; i < 8; ++i) d[i] = 0u;
                }
            }

            // ---- stage B tile: 128 cols x 32 k bf16 via Tensor Data Mover ----
#if HAVE_TDM
            if (wave == 0) {
                tdm_load_b_tile(WtCol + k0, bs_lds);
                __builtin_amdgcn_s_wait_tensorcnt(0);
            }
#else
            {
                const uint4* p = (const uint4*)(WtCol + (size_t)sRow * KTOT + k0 + sK);
                const uint4 u0 = p[0];
                const uint4 u1 = p[1];
                *(uint4*)&Bs[sRow][sK]     = u0;
                *(uint4*)&Bs[sRow][sK + 8] = u1;
            }
#endif
            __syncthreads();

            v16bf bfr[2], afr[4];
#pragma unroll
            for (int nt = 0; nt < 2; ++nt)
                bfr[nt] = load_frag(&Bs[nBase + nt * 16 + lr][0], kh);
#pragma unroll
            for (int mt = 0; mt < 4; ++mt)
                afr[mt] = load_frag(&As[mBase + mt * 16 + lr][0], kh);

#pragma unroll
            for (int mt = 0; mt < 4; ++mt)
#pragma unroll
                for (int nt = 0; nt < 2; ++nt)
                    acc[mt][nt] = __builtin_amdgcn_wmma_f32_16x16x32_bf16(
                        false, afr[mt], false, bfr[nt], (short)0, acc[mt][nt], false, false);
            __syncthreads();
        }
    }

    // ---- epilogue: C/D layout (lane,VGPR)->(M,N); ReLU (alpha) + BN affine ----
#pragma unroll
    for (int mt = 0; mt < 4; ++mt) {
#pragma unroll
        for (int nt = 0; nt < 2; ++nt) {
            const int gn = colBlock + nBase + nt * 16 + lr;
            const float s  = bnscale[gn];
            const float sh = bnshift[gn];
            const bool isAlpha = (gn < 256);
#pragma unroll
            for (int v = 0; v < 8; ++v) {
                const int gm = rowBlock + mBase + mt * 16 + ((lane < 16) ? v : v + 8);
                if (gm < NNODES) {
                    float cv = acc[mt][nt][v];
                    if (isAlpha) cv = fmaxf(cv, 0.0f);
                    out[(size_t)gm * COUT + gn] = cv * s + sh;
                }
            }
        }
    }
}

// ---------------------------------------------------------------------------
extern "C" void kernel_launch(void* const* d_in, const int* in_sizes, int n_in,
                              void* d_out, int out_size, void* d_ws, size_t ws_size,
                              hipStream_t stream) {
    (void)in_sizes; (void)n_in; (void)out_size; (void)ws_size;

    const float* x      = (const float*)d_in[0];
    const int*   erow   = (const int*)d_in[1];
    const int*   ecol   = (const int*)d_in[2];
    const float* eval   = (const float*)d_in[3];
    const float* wa_x   = (const float*)d_in[4];
    const float* wa_adj = (const float*)d_in[5];
    const float* wb_x   = (const float*)d_in[6];
    const float* wb_adj = (const float*)d_in[7];
    const float* bnag = (const float*)d_in[8];
    const float* bnab = (const float*)d_in[9];
    const float* bnam = (const float*)d_in[10];
    const float* bnav = (const float*)d_in[11];
    const float* bnbg = (const float*)d_in[12];
    const float* bnbb = (const float*)d_in[13];
    const float* bnbm = (const float*)d_in[14];
    const float* bnbv = (const float*)d_in[15];
    float* out = (float*)d_out;

    const size_t NF = (size_t)NNODES * FIN;
    float* ws   = (float*)d_ws;
    float* hop0 = ws;
    float* hop1 = ws + NF;
    float* hop2 = ws + 2 * NF;
    float* tmp  = ws + 3 * NF;
    __bf16* Wt  = (__bf16*)(ws + 4 * NF);
    float* bnscale = (float*)(Wt + (size_t)COUT * KTOT);
    float* bnshift = bnscale + COUT;

    // zero the 4 atomic-accumulation buffers
    hipMemsetAsync(ws, 0, 4 * NF * sizeof(float), stream);

    prep_kernel<<<dim3(COUT, 4), 256, 0, stream>>>(
        wa_x, wa_adj, wb_x, wb_adj,
        bnag, bnab, bnam, bnav, bnbg, bnbb, bnbm, bnbv,
        Wt, bnscale, bnshift);

    const int spmmBlocks = (NEDGES + 7) / 8;   // 8 waves (edges) per block
    spmm_kernel<<<spmmBlocks, 256, 0, stream>>>(x,    hop0, erow, ecol, eval);
    spmm_kernel<<<spmmBlocks, 256, 0, stream>>>(hop0, hop1, erow, ecol, eval);
    spmm_kernel<<<spmmBlocks, 256, 0, stream>>>(hop1, tmp,  erow, ecol, eval);
    spmm_kernel<<<spmmBlocks, 256, 0, stream>>>(tmp,  hop2, erow, ecol, eval);

    gemm_kernel<<<dim3((NNODES + BM - 1) / BM, COUT / BN), 256, 0, stream>>>(
        x, hop0, hop1, hop2, Wt, bnscale, bnshift, out);
}
